// inter_frame_attention_2267742732706
// MI455X (gfx1250) — matrix operand order repaired and checked by reading, not compiled, not run on hardware
//
#include <hip/hip_runtime.h>
#include <hip/hip_bf16.h>

#define B_ 4
#define N_ 512
#define D_ 128
#define O_ 64
#define H_ 4

typedef __attribute__((ext_vector_type(16))) __bf16          v16bf;
typedef __attribute__((ext_vector_type(8)))  float           v8f;
typedef __attribute__((ext_vector_type(16))) unsigned short  v16u;
typedef __attribute__((ext_vector_type(8)))  unsigned short  v8u;

// float -> bf16, round-to-nearest-even
static __device__ __forceinline__ unsigned short f2bf(float f) {
    unsigned u = __float_as_uint(f);
    unsigned r = u + 0x7FFFu + ((u >> 16) & 1u);
    return (unsigned short)(r >> 16);
}

static __device__ __forceinline__ v16bf as_bf16(v16u v) {
    return __builtin_bit_cast(v16bf, v);
}

// gfx1250 has a hardware V_TANH_F32 transcendental — use it if clang exposes it.
static __device__ __forceinline__ float dev_tanh(float x) {
#if __has_builtin(__builtin_amdgcn_tanhf)
    return __builtin_amdgcn_tanhf(x);
#else
    return tanhf(x);
#endif
}

// ---------------------------------------------------------------------------
// Kernel 0: x (f32 [B,N,D]) -> bf16 row-major x_bf[b][j][d] and bf16
// transposed xT_bf[b][d][j] in workspace.
// ---------------------------------------------------------------------------
__global__ void k_convert(const float* __restrict__ x,
                          unsigned short* __restrict__ x_bf,
                          unsigned short* __restrict__ xT_bf) {
    int idx = blockIdx.x * 256 + threadIdx.x;          // B*N*D = 262144
    if (idx >= B_ * N_ * D_) return;
    int d = idx & (D_ - 1);
    int j = (idx / D_) & (N_ - 1);
    int b = idx / (D_ * N_);
    unsigned short v = f2bf(x[idx]);
    x_bf[idx] = v;
    xT_bf[(b * D_ + d) * N_ + j] = v;
}

// ---------------------------------------------------------------------------
// Kernel 1: one workgroup per (b, i). 256 threads = 8 wave32s.
//  Phase 1: W'T[o][d] = bf16(x[b,i,d] * att_proj_w[d,o])  (LDS, 16KB)
//  Phase 2: att scores via WMMA GEMM [512j x 128d] @ [128d x 64o],
//           hardware tanh, contract with att_weight (O->H), cross-lane reduce.
//  Phase 3: softmax over j per head; amT[16][512] bf16 in LDS (rows 4..15 = 0)
//  Phase 4: x1[d,h] via WMMA GEMM  xT[128d x 512j] @ am[512j x 16h]
//  Phase 5: y[o] = x1 @ proj_att_w + x_i @ proj_noatt_w + biases  -> y_ws
// ---------------------------------------------------------------------------
__global__ void __launch_bounds__(256, 1)
k_attention(const float* __restrict__ x,
            const float* __restrict__ att_proj_w,   // [D,O]
            const float* __restrict__ att_proj_b,   // [O]
            const float* __restrict__ att_weight,   // [O,H]
            const float* __restrict__ proj_att_w,   // [D*H,O]
            const float* __restrict__ proj_att_b,   // [O]
            const float* __restrict__ proj_noatt_w, // [D,O]
            const float* __restrict__ proj_noatt_b, // [O]
            const unsigned short* __restrict__ x_bf,
            const unsigned short* __restrict__ xT_bf,
            float* __restrict__ y_ws) {
    __shared__ unsigned short wpT[O_][D_];    // 16 KB  (W' transposed)
    __shared__ float          scores[N_][H_]; //  8 KB  (pre-softmax -> exp)
    __shared__ unsigned short amT[16][N_];    // 16 KB  (softmax, bf16, padded)
    __shared__ float          x1s[D_ * H_];   //  2 KB
    __shared__ float          red256[256];    //  1 KB
    __shared__ float          mx[H_], sm[H_];

    const int tid  = threadIdx.x;
    const int wave = tid >> 5;
    const int lane = tid & 31;
    const int l16  = lane & 15;
    const int lh   = lane >> 4;

    const int b = blockIdx.x / N_;
    const int i = blockIdx.x - b * N_;
    const float* xi = x + (size_t)(b * N_ + i) * D_;

    // ---- Phase 1: W'T in LDS ----
    for (int idx = tid; idx < O_ * D_; idx += 256) {
        int o = idx >> 7;            // /128
        int d = idx & (D_ - 1);
        wpT[o][d] = f2bf(xi[d] * att_proj_w[d * O_ + o]);
    }
    __syncthreads();

    // ---- Phase 2: att GEMM + tanh + O->H contraction ----
    // Preload all 16 B fragments of W' (k-step x o-tile): 32B contiguous LDS.
    v16u bfrag[4][4];
#pragma unroll
    for (int k = 0; k < 4; ++k)
#pragma unroll
        for (int ot = 0; ot < 4; ++ot)
            bfrag[k][ot] = *(const v16u*)&wpT[ot * 16 + l16][k * 32 + lh * 16];

    float aw_l[4][H_];
#pragma unroll
    for (int ot = 0; ot < 4; ++ot)
#pragma unroll
        for (int h = 0; h < H_; ++h)
            aw_l[ot][h] = att_weight[(ot * 16 + l16) * H_ + h];

    const unsigned short* xb = x_bf + (size_t)b * N_ * D_; // [N][D]

    for (int jt = wave; jt < 32; jt += 8) {
        v8f c0 = {}, c1 = {}, c2 = {}, c3 = {};
#pragma unroll
        for (int k = 0; k < 4; ++k) {
            // A fragment: row j = jt*16 + l16
            // halves 0-7: d = k*32 + lh*16 ; halves 8-15: d = k*32 + lh*16 + 8
            const unsigned short* pa = xb + (jt * 16 + l16) * D_ + k * 32 + lh * 16;
            v8u a0 = *(const v8u*)pa;
            v8u a1 = *(const v8u*)(pa + 8);
            v16u a;
#pragma unroll
            for (int e = 0; e < 8; ++e) { a[e] = a0[e]; a[8 + e] = a1[e]; }
            v16bf A = as_bf16(a);
            c0 = __builtin_amdgcn_wmma_f32_16x16x32_bf16(false, A, false, as_bf16(bfrag[k][0]), (short)0, c0, false, false);
            c1 = __builtin_amdgcn_wmma_f32_16x16x32_bf16(false, A, false, as_bf16(bfrag[k][1]), (short)0, c1, false, false);
            c2 = __builtin_amdgcn_wmma_f32_16x16x32_bf16(false, A, false, as_bf16(bfrag[k][2]), (short)0, c2, false, false);
            c3 = __builtin_amdgcn_wmma_f32_16x16x32_bf16(false, A, false, as_bf16(bfrag[k][3]), (short)0, c3, false, false);
        }
        // tanh(att + bias) then per-lane partial of scores[j][h]
        float part[8][H_];
#pragma unroll
        for (int r = 0; r < 8; ++r)
#pragma unroll
            for (int h = 0; h < H_; ++h) part[r][h] = 0.f;
#pragma unroll
        for (int ot = 0; ot < 4; ++ot) {
            float bias = att_proj_b[ot * 16 + l16];
            v8f cc = (ot == 0) ? c0 : (ot == 1) ? c1 : (ot == 2) ? c2 : c3;
#pragma unroll
            for (int r = 0; r < 8; ++r) {
                float t = dev_tanh(cc[r] + bias);
#pragma unroll
                for (int h = 0; h < H_; ++h) part[r][h] += t * aw_l[ot][h];
            }
        }
        // reduce over the 16 lanes of each half (o dimension)
#pragma unroll
        for (int m = 1; m < 16; m <<= 1)
#pragma unroll
            for (int r = 0; r < 8; ++r)
#pragma unroll
                for (int h = 0; h < H_; ++h)
                    part[r][h] += __shfl_xor(part[r][h], m);
        if (l16 == 0) {
            int jb = jt * 16 + lh * 8;   // C layout: M = r + 8*lh
#pragma unroll
            for (int r = 0; r < 8; ++r)
#pragma unroll
                for (int h = 0; h < H_; ++h)
                    scores[jb + r][h] = part[r][h];
        }
    }
    __syncthreads();

    // ---- Phase 3: softmax over j per head ----
    {
        int h = tid & 3, g = tid >> 2;        // 64 groups x 8 j
        float m = -1e30f;
        for (int r = 0; r < 8; ++r) m = fmaxf(m, scores[g * 8 + r][h]);
        red256[tid] = m;
        __syncthreads();
        if (tid < H_) {
            float mm = -1e30f;
            for (int gg = 0; gg < 64; ++gg) mm = fmaxf(mm, red256[gg * 4 + tid]);
            mx[tid] = mm;
        }
        __syncthreads();
        float s = 0.f;
        float mh = mx[h];
        for (int r = 0; r < 8; ++r) {
            float e = __expf(scores[g * 8 + r][h] - mh);
            scores[g * 8 + r][h] = e;       // cache exp; sole writer of this slot
            s += e;
        }
        red256[tid] = s;
        __syncthreads();
        if (tid < H_) {
            float ss = 0.f;
            for (int gg = 0; gg < 64; ++gg) ss += red256[gg * 4 + tid];
            sm[tid] = ss;
        }
        __syncthreads();
        float rs[H_];
#pragma unroll
        for (int hh = 0; hh < H_; ++hh) rs[hh] = 1.f / sm[hh];
        for (int idx = tid; idx < 16 * N_; idx += 256) {
            int hh = idx / N_, j = idx & (N_ - 1);
            unsigned short v = 0;
            if (hh < H_) v = f2bf(scores[j][hh] * rs[hh]);
            amT[hh][j] = v;
        }
    }
    __syncthreads();

    // ---- Phase 4: x1[d,h] = sum_j xT[d,j]*am[j,h] via WMMA ----
    {
        const unsigned short* xtb = xT_bf + (size_t)b * D_ * N_; // [D][N]
        const int d0 = wave * 16;
        v8f c = {};
#pragma unroll
        for (int ks = 0; ks < 16; ++ks) {
            int j0 = ks * 32;
            const unsigned short* pa = xtb + (d0 + l16) * N_ + j0 + lh * 16;
            v8u a0 = *(const v8u*)pa;
            v8u a1 = *(const v8u*)(pa + 8);
            v16u a;
#pragma unroll
            for (int e = 0; e < 8; ++e) { a[e] = a0[e]; a[8 + e] = a1[e]; }
            v16u bb = *(const v16u*)&amT[l16][j0 + lh * 16];
            c = __builtin_amdgcn_wmma_f32_16x16x32_bf16(false, as_bf16(a), false, as_bf16(bb), (short)0, c, false, false);
        }
        if (l16 < H_) {
#pragma unroll
            for (int r = 0; r < 8; ++r)
                x1s[(d0 + r + 8 * lh) * H_ + l16] = c[r];  // f = d*H + h
        }
    }
    __syncthreads();

    // ---- Phase 5: output projections ----
    {
        int o = tid & 63, g = tid >> 6;
        float acc = 0.f;
        for (int f = g * 128; f < g * 128 + 128; ++f)
            acc += x1s[f] * proj_att_w[f * O_ + o];
        for (int d = g * 32; d < g * 32 + 32; ++d)
            acc += xi[d] * proj_noatt_w[d * O_ + o];
        red256[tid] = acc;
        __syncthreads();
        if (tid < 64) {
            float y = red256[tid] + red256[64 + tid] + red256[128 + tid] + red256[192 + tid]
                    + proj_att_b[tid] + proj_noatt_b[tid];
            y_ws[(size_t)(b * N_ + i) * O_ + tid] = y;
        }
    }
}

// ---------------------------------------------------------------------------
// Kernel 2a: deterministic partial BN stats. 32 blocks, block bk owns rows
// [bk*64, bk*64+64). stats[bk][0][o]=sum, stats[bk][1][o]=sumsq.
// ---------------------------------------------------------------------------
__global__ void k_bn_stats(const float* __restrict__ y_ws,
                           float* __restrict__ stats) {
    __shared__ float s1[256], s2[256];
    const int tid = threadIdx.x;
    const int o = tid & 63, g = tid >> 6;
    const int r0 = blockIdx.x * 64;
    float a = 0.f, b2 = 0.f;
    for (int r = r0 + g; r < r0 + 64; r += 4) {
        float v = y_ws[r * O_ + o];
        a += v; b2 += v * v;
    }
    s1[tid] = a; s2[tid] = b2;
    __syncthreads();
    if (tid < O_) {
        stats[blockIdx.x * 128 + tid]      = s1[tid] + s1[64 + tid] + s1[128 + tid] + s1[192 + tid];
        stats[blockIdx.x * 128 + 64 + tid] = s2[tid] + s2[64 + tid] + s2[128 + tid] + s2[192 + tid];
    }
}

// ---------------------------------------------------------------------------
// Kernel 2b: finish stats (each block re-reduces the 32 partials — 16KB, cheap
// and bitwise-deterministic) then apply BN (training stats) + SELU.
// 128 blocks x 1024 elements.
// ---------------------------------------------------------------------------
__global__ void k_bn_apply(const float* __restrict__ y_ws,
                           const float* __restrict__ stats,
                           const float* __restrict__ gamma,
                           const float* __restrict__ beta,
                           float* __restrict__ out) {
    __shared__ float mu[O_], inv[O_];
    const int tid = threadIdx.x;
    const float Mf = (float)(B_ * N_);
    if (tid < O_) {
        float S = 0.f, S2 = 0.f;
        for (int p = 0; p < 32; ++p) {
            S  += stats[p * 128 + tid];
            S2 += stats[p * 128 + 64 + tid];
        }
        float m   = S / Mf;
        float var = S2 / Mf - m * m;            // biased variance
        mu[tid]  = m;
        inv[tid] = rsqrtf(var + 1e-5f);
    }
    __syncthreads();
    const float SC = 1.0507009873554804934f, AL = 1.6732632423543772848f;
    const int base = blockIdx.x * 1024;
    for (int idx = base + tid; idx < base + 1024; idx += 256) {
        int oo = idx & 63;
        float v = (y_ws[idx] - mu[oo]) * inv[oo] * gamma[oo] + beta[oo];
        out[idx] = v > 0.f ? SC * v : SC * AL * (__expf(v) - 1.f);
    }
}

extern "C" void kernel_launch(void* const* d_in, const int* in_sizes, int n_in,
                              void* d_out, int out_size, void* d_ws, size_t ws_size,
                              hipStream_t stream) {
    (void)in_sizes; (void)n_in; (void)out_size; (void)ws_size;
    const float* x            = (const float*)d_in[0];
    const float* att_proj_w   = (const float*)d_in[1];
    const float* att_proj_b   = (const float*)d_in[2];
    const float* att_weight   = (const float*)d_in[3];
    const float* proj_att_w   = (const float*)d_in[4];
    const float* proj_att_b   = (const float*)d_in[5];
    const float* proj_noatt_w = (const float*)d_in[6];
    const float* proj_noatt_b = (const float*)d_in[7];
    const float* bn_gamma     = (const float*)d_in[8];
    const float* bn_beta      = (const float*)d_in[9];

    unsigned short* x_bf  = (unsigned short*)d_ws;             // 512 KB
    unsigned short* xT_bf = x_bf + (size_t)B_ * N_ * D_;       // 512 KB
    float*          y_ws  = (float*)(xT_bf + (size_t)B_ * N_ * D_); // 512 KB
    float*          stats = y_ws + (size_t)B_ * N_ * O_;       // 16 KB

    k_convert<<<(B_ * N_ * D_) / 256, 256, 0, stream>>>(x, x_bf, xT_bf);
    k_attention<<<B_ * N_, 256, 0, stream>>>(x, att_proj_w, att_proj_b, att_weight,
                                             proj_att_w, proj_att_b,
                                             proj_noatt_w, proj_noatt_b,
                                             x_bf, xT_bf, y_ws);
    k_bn_stats<<<32, 256, 0, stream>>>(y_ws, stats);
    k_bn_apply<<<128, 256, 0, stream>>>(y_ws, stats, bn_gamma, bn_beta, (float*)d_out);
}
